// CrossAttention_73847667688170
// MI455X (gfx1250) — matrix operand order repaired and checked by reading, not compiled, not run on hardware
//
#include <hip/hip_runtime.h>
#include <stdint.h>

// Cross-attention (B=8, C=256, H=W=64, N=4096), fp32 in/out.
// bf16 WMMA (v_wmma_f32_16x16x32_bf16) for projections + both attention GEMMs,
// f32 accumulate, flash-attention online softmax.
// CDNA5 paths: WMMA, double-buffered Tensor Data Mover staging
// (tensor_load_to_lds + s_wait_tensorcnt), s_wait_dscnt, native bf16 cvt.
// Workspace: Qt/Kt [B][N][C] bf16, Vm [B][C][N] bf16 (48 MB) + 3 pre-swizzled
// weight images (384 KB).

#define B_ 8
#define C_ 256
#define N_ 4096

typedef __attribute__((ext_vector_type(16))) __bf16       v16bf;
typedef __attribute__((ext_vector_type(8)))  float        v8f;
typedef __attribute__((ext_vector_type(8)))  unsigned int v8u;
typedef __attribute__((ext_vector_type(4)))  unsigned int u32x4;
typedef __attribute__((ext_vector_type(8)))  int          i32x8;
typedef __attribute__((ext_vector_type(4)))  int          i32x4;

// Native f32 -> bf16 (v_cvt_pk_bf16_f32 when paired)
__device__ __forceinline__ __bf16 f2bf(float f) { return (__bf16)f; }
__device__ __forceinline__ unsigned int pack2(float a, float b) {
    __bf16 x = (__bf16)a, y = (__bf16)b;
    unsigned short lo, hi;
    __builtin_memcpy(&lo, &x, 2);
    __builtin_memcpy(&hi, &y, 2);
    return (unsigned int)lo | ((unsigned int)hi << 16);
}
// ISA 7.12.2: 16-bit A-matrix 16x32. Lane<16 holds K {0..7,16..23}, lane>=16 {8..15,24..31}.
__device__ __forceinline__ int aoff(int lane, int v) {
    return ((lane < 16) ? 0 : 8) + ((v < 4) ? 2 * v : 16 + 2 * (v - 4));
}
// 16-bit B-matrix 32x16: lane = column (mod 16), lanes 0-15 hold K {0..7,16..23}, 16-31 hold K {8..15,24..31}.
__device__ __forceinline__ int boff(int lane, int v) {
    return ((lane < 16) ? 0 : 8) + ((v < 4) ? 2 * v : 16 + 2 * (v - 4));
}

// ---------------------------------------------------------------------------
// Tensor Data Mover: 2D tile load (bf16 elements) into LDS.
// Descriptor layout per CDNA5 ISA §8.3 (group0) / §8.4 (group1).
// This toolchain uses the 6-arg builtin: (g0, g1, g2, g3, g_extra, cpol).
// ---------------------------------------------------------------------------
__device__ __forceinline__ void tdm_load_2d_bf16(
    unsigned int lds_off, const void* gaddr,
    unsigned int tensor_d0, unsigned int tensor_d1,
    unsigned int tile_d0, unsigned int tile_d1,
    unsigned long long stride0 /* elements */)
{
    unsigned long long ga = (unsigned long long)(uintptr_t)gaddr;
    u32x4 g0;
    g0[0] = 1u;                                        // count=1 (valid), user D#
    g0[1] = lds_off;                                   // lds_addr [63:32]
    g0[2] = (unsigned int)(ga & 0xFFFFFFFFu);          // global_addr[31:0]
    g0[3] = (unsigned int)((ga >> 32) & 0x01FFFFFFu)   // global_addr[56:32]
          | (2u << 30);                                // type=2 ("image")
    i32x8 g1;
    g1[0] = (int)(1u << 16);                           // data_size=1 (2 bytes), wg_mask=0
    g1[1] = (int)((tensor_d0 & 0xFFFFu) << 16);        // tensor_dim0[15:0]
    g1[2] = (int)(((tensor_d0 >> 16) & 0xFFFFu)        // tensor_dim0[31:16]
          | ((tensor_d1 & 0xFFFFu) << 16));            // tensor_dim1[15:0]
    g1[3] = (int)(((tensor_d1 >> 16) & 0xFFFFu)        // tensor_dim1[31:16]
          | ((tile_d0 & 0xFFFFu) << 16));              // tile_dim0
    g1[4] = (int)(tile_d1 & 0xFFFFu);                  // tile_dim1 (tile_dim2=0)
    g1[5] = (int)(stride0 & 0xFFFFFFFFu);              // tensor_dim0_stride[31:0]
    g1[6] = (int)((stride0 >> 32) & 0xFFFFu);          // stride[47:32] (dim1_stride=0)
    g1[7] = 0;
    i32x4 z4 = {0, 0, 0, 0};
    i32x8 z8 = {0, 0, 0, 0, 0, 0, 0, 0};
    __builtin_amdgcn_tensor_load_to_lds(g0, g1, z4, z4, z8, 0);
}
__device__ __forceinline__ unsigned int lds_off_of(const void* p) {
    // Generic pointer to LDS: low 32 bits are the LDS byte address (ISA §10.2)
    return (unsigned int)(uintptr_t)p;
}

// ---------------------------------------------------------------------------
// One-shot W pre-swizzle: pack W (f32 [C][C]) into A-fragment register order,
// Wsw[((ot*8 + kk)*32 + lane)*8 + v] = pack2(W[o][k], W[o][k+1])
// with o = ot*16 + (lane&15), k = kk*32 + aoff(lane, v).
// Proj A-loads then become one aligned 32B load, zero conversion VALU.
// ---------------------------------------------------------------------------
__global__ __launch_bounds__(128) void wswz_kernel(
    const float* __restrict__ W, unsigned int* __restrict__ Wsw)
{
    int idx  = blockIdx.x * 128 + threadIdx.x;   // 0 .. 32767
    int v    = idx & 7;
    int lane = (idx >> 3) & 31;
    int kk   = (idx >> 8) & 7;
    int ot   = idx >> 11;
    int o    = ot * 16 + (lane & 15);
    int k    = kk * 32 + aoff(lane, v);
    Wsw[idx] = pack2(W[(size_t)o * C_ + k], W[(size_t)o * C_ + k + 1]);
}

// ---------------------------------------------------------------------------
// Projection: Y[b][o][n] = sum_c W[o][c] * x[b][c][n] + bias[o]
// One wave per (16-column n-tile, batch); 16 o-tiles x 8 k-steps = 128 WMMAs.
// Per o-tile: preload all 8 W-frags (one load clause), then 8 chained WMMAs.
// TRANSPOSED==1 -> store Y^T as out[b][n][o]  (for Q, K)
// TRANSPOSED==0 -> store Y   as out[b][o][n]  (for V)
// ---------------------------------------------------------------------------
template <int TRANSPOSED>
__global__ __launch_bounds__(32) void proj_kernel(
    const float* __restrict__ x, const unsigned int* __restrict__ Wsw,
    const float* __restrict__ bias, __bf16* __restrict__ outp)
{
    const int lane = threadIdx.x;
    const int n0   = blockIdx.x * 16;
    const int b    = blockIdx.y;
    const float* xb = x + (size_t)b * C_ * N_;
    const int rowh = (lane < 16) ? 0 : 8;
    const int ncol = n0 + (lane & 15);
    const v8u* wswv = (const v8u*)Wsw;

    // Hoist B-frags (X tile; 16 lanes read consecutive n -> coalesced)
    v16bf xfrag[8];
    #pragma unroll
    for (int kk = 0; kk < 8; ++kk) {
        v8u u;
        #pragma unroll
        for (int v = 0; v < 8; ++v) {
            int c = kk * 32 + boff(lane, v);
            u[v] = pack2(xb[(size_t)c * N_ + ncol], xb[(size_t)(c + 1) * N_ + ncol]);
        }
        xfrag[kk] = __builtin_bit_cast(v16bf, u);
    }

    for (int ot = 0; ot < 16; ++ot) {
        const int o0 = ot * 16;

        // Preload the 8 A-frags for this o-tile (pure b128 loads, no cvt)
        v8u wf[8];
        #pragma unroll
        for (int kk = 0; kk < 8; ++kk)
            wf[kk] = wswv[(ot * 8 + kk) * 32 + lane];

        v8f acc;
        #pragma unroll
        for (int r = 0; r < 8; ++r) acc[r] = bias[o0 + rowh + r];

        #pragma unroll
        for (int kk = 0; kk < 8; ++kk) {
            v16bf af = __builtin_bit_cast(v16bf, wf[kk]);
            acc = __builtin_amdgcn_wmma_f32_16x16x32_bf16(
                false, af, false, xfrag[kk], (short)0, acc, false, false);
        }

        if (TRANSPOSED) {
            // Qt[b][n][o]: lane writes 8 consecutive bf16 -> one b128 store
            __bf16* dst = outp + ((size_t)b * N_ + n0 + (lane & 15)) * C_ + o0 + rowh;
            uint4 pk;
            pk.x = pack2(acc[0], acc[1]); pk.y = pack2(acc[2], acc[3]);
            pk.z = pack2(acc[4], acc[5]); pk.w = pack2(acc[6], acc[7]);
            *(uint4*)dst = pk;
        } else {
            // Vm[b][o][n]: coalesced across lanes per r
            __bf16* dst = outp + ((size_t)b * C_ + o0 + rowh) * (size_t)N_ + n0 + (lane & 15);
            #pragma unroll
            for (int r = 0; r < 8; ++r) dst[(size_t)r * N_] = f2bf(acc[r]);
        }
    }
}

// ---------------------------------------------------------------------------
// Flash attention: block = 128 threads (4 waves), each wave owns 16 queries.
// TDM double-buffers K/V tiles in LDS (overlap DMA with WMMA); per 32-key
// step: S = Q^T K (16 WMMAs), online softmax, O += P V (16 WMMAs).
// ---------------------------------------------------------------------------
__global__ __launch_bounds__(128) void attn_kernel(
    const __bf16* __restrict__ Qt,  // [B][N][C]
    const __bf16* __restrict__ Kt,  // [B][N][C]
    const __bf16* __restrict__ Vm,  // [B][C][N]
    float* __restrict__ out)        // [B][C][N]
{
    __shared__ __bf16 Qs[64 * C_];         // [n_local][c]       32 KB
    __shared__ __bf16 Ks[2][32 * C_];      // [buf][m_local][c]  2x16 KB
    __shared__ __bf16 Vs[2][C_ * 32];      // [buf][c][m_local]  2x16 KB
    __shared__ __bf16 Ps[4 * 16 * 32];     // per-wave P tile    4 KB

    const int tid  = threadIdx.x;
    const int lane = tid & 31;
    const int wave = tid >> 5;
    const int nblk = blockIdx.x * 64;
    const int b    = blockIdx.y;
    const int rowh = (lane < 16) ? 0 : 8;

    // ---- TDM: stage Q chunk [64 n][256 c] + first K/V tiles ----
    if (tid < 32) {
        tdm_load_2d_bf16(lds_off_of(Qs), Qt + ((size_t)b * N_ + nblk) * C_,
                         C_, N_, C_, 64, C_);
        tdm_load_2d_bf16(lds_off_of(Ks[0]), Kt + (size_t)b * N_ * C_,
                         C_, N_, C_, 32, C_);
        tdm_load_2d_bf16(lds_off_of(Vs[0]), Vm + (size_t)b * C_ * N_,
                         N_, C_, 32, C_, N_);
        __builtin_amdgcn_s_wait_tensorcnt(2);   // Q done (in-order)
    }
    __syncthreads();

    // Hoist Q A-frags (16 rows x 256 ch = 64 VGPRs/lane), loaded once
    v16bf qfrag[8];
    {
        const int nloc = wave * 16 + (lane & 15);
        #pragma unroll
        for (int kk = 0; kk < 8; ++kk) {
            v8u u;
            #pragma unroll
            for (int v = 0; v < 8; ++v) {
                int k = kk * 32 + aoff(lane, v);   // even -> b32 load of 2 bf16
                u[v] = *(const unsigned int*)&Qs[nloc * C_ + k];
            }
            qfrag[kk] = __builtin_bit_cast(v16bf, u);
        }
    }

    float m_run[8], l_run[8];
    v8f oacc[16];
    #pragma unroll
    for (int r = 0; r < 8; ++r) { m_run[r] = -1.0e30f; l_run[r] = 0.0f; }
    #pragma unroll
    for (int ct = 0; ct < 16; ++ct)
        #pragma unroll
        for (int r = 0; r < 8; ++r) oacc[ct][r] = 0.0f;

    for (int step = 0; step < N_ / 32; ++step) {
        const int m0  = step * 32;
        const int cur = step & 1;
        __syncthreads();   // everyone done reading buffer cur^1 (step-1)
        if (tid < 32) {
            if (m0 + 32 < N_) {
                // Issue next tile pair into the other buffer, then wait until
                // only those 2 remain outstanding => current pair complete.
                tdm_load_2d_bf16(lds_off_of(Ks[cur ^ 1]),
                                 Kt + ((size_t)b * N_ + m0 + 32) * C_,
                                 C_, N_, C_, 32, C_);
                tdm_load_2d_bf16(lds_off_of(Vs[cur ^ 1]),
                                 Vm + (size_t)b * C_ * N_ + m0 + 32,
                                 N_, C_, 32, C_, N_);
                __builtin_amdgcn_s_wait_tensorcnt(2);
            } else {
                __builtin_amdgcn_s_wait_tensorcnt(0);
            }
        }
        __syncthreads();
        const __bf16* ks = Ks[cur];
        const __bf16* vs = Vs[cur];

        // ---- S = Q^T K : two 16-column halves, 8 k-steps each ----
        v8f s0, s1;
        #pragma unroll
        for (int r = 0; r < 8; ++r) { s0[r] = 0.0f; s1[r] = 0.0f; }
        const int mcol = lane & 15;
        #pragma unroll
        for (int kk = 0; kk < 8; ++kk) {
            v8u u0, u1;
            #pragma unroll
            for (int v = 0; v < 8; ++v) {
                int c = kk * 32 + boff(lane, v);
                u0[v] = *(const unsigned int*)&ks[mcol * C_ + c];
                u1[v] = *(const unsigned int*)&ks[(mcol + 16) * C_ + c];
            }
            v16bf kf0 = __builtin_bit_cast(v16bf, u0);
            v16bf kf1 = __builtin_bit_cast(v16bf, u1);
            s0 = __builtin_amdgcn_wmma_f32_16x16x32_bf16(
                false, qfrag[kk], false, kf0, (short)0, s0, false, false);
            s1 = __builtin_amdgcn_wmma_f32_16x16x32_bf16(
                false, qfrag[kk], false, kf1, (short)0, s1, false, false);
        }

        // ---- online softmax over the 32 keys of this step ----
        float scale_r[8];
        #pragma unroll
        for (int r = 0; r < 8; ++r) {
            float mx = fmaxf(s0[r], s1[r]);
            mx = fmaxf(mx, __shfl_xor(mx, 1, 32));
            mx = fmaxf(mx, __shfl_xor(mx, 2, 32));
            mx = fmaxf(mx, __shfl_xor(mx, 4, 32));
            mx = fmaxf(mx, __shfl_xor(mx, 8, 32));
            float mnew = fmaxf(m_run[r], mx);
            float p0 = __expf(s0[r] - mnew);
            float p1 = __expf(s1[r] - mnew);
            s0[r] = p0; s1[r] = p1;
            float sum = p0 + p1;
            sum += __shfl_xor(sum, 1, 32);
            sum += __shfl_xor(sum, 2, 32);
            sum += __shfl_xor(sum, 4, 32);
            sum += __shfl_xor(sum, 8, 32);
            float sc = __expf(m_run[r] - mnew);
            l_run[r] = l_run[r] * sc + sum;
            m_run[r] = mnew;
            scale_r[r] = sc;
        }
        #pragma unroll
        for (int ct = 0; ct < 16; ++ct)
            #pragma unroll
            for (int r = 0; r < 8; ++r) oacc[ct][r] *= scale_r[r];

        // ---- transpose P (D layout -> A layout) through per-wave LDS ----
        __bf16* ps = Ps + wave * (16 * 32);
        #pragma unroll
        for (int r = 0; r < 8; ++r) {
            ps[(r + rowh) * 32 + mcol]      = f2bf(s0[r]);
            ps[(r + rowh) * 32 + 16 + mcol] = f2bf(s1[r]);
        }
        asm volatile("s_wait_dscnt 0" ::: "memory");  // same-wave DS RAW
        v16bf pf;
        {
            v8u u;
            #pragma unroll
            for (int v = 0; v < 8; ++v)
                u[v] = *(const unsigned int*)&ps[(lane & 15) * 32 + aoff(lane, v)];
            pf = __builtin_bit_cast(v16bf, u);
        }

        // ---- O += P x V : 16 channel tiles ----
        #pragma unroll
        for (int ct = 0; ct < 16; ++ct) {
            v8u u;
            const int ccol = ct * 16 + (lane & 15);
            #pragma unroll
            for (int v = 0; v < 8; ++v)
                u[v] = *(const unsigned int*)&vs[ccol * 32 + boff(lane, v)];
            v16bf vf = __builtin_bit_cast(v16bf, u);
            oacc[ct] = __builtin_amdgcn_wmma_f32_16x16x32_bf16(
                false, pf, false, vf, (short)0, oacc[ct], false, false);
        }
    }

    // ---- normalize and store: out[b][c][n], 8 consecutive n per lane ----
    float linv[8];
    #pragma unroll
    for (int r = 0; r < 8; ++r) linv[r] = 1.0f / l_run[r];
    #pragma unroll
    for (int ct = 0; ct < 16; ++ct) {
        const int c = ct * 16 + (lane & 15);
        float* dst = out + ((size_t)b * C_ + c) * N_ + nblk + wave * 16 + rowh;
        float4 lo = make_float4(oacc[ct][0] * linv[0], oacc[ct][1] * linv[1],
                                oacc[ct][2] * linv[2], oacc[ct][3] * linv[3]);
        float4 hi = make_float4(oacc[ct][4] * linv[4], oacc[ct][5] * linv[5],
                                oacc[ct][6] * linv[6], oacc[ct][7] * linv[7]);
        *(float4*)dst = lo;
        *(float4*)(dst + 4) = hi;
    }
}

extern "C" void kernel_launch(void* const* d_in, const int* in_sizes, int n_in,
                              void* d_out, int out_size, void* d_ws, size_t ws_size,
                              hipStream_t stream) {
    const float* x1 = (const float*)d_in[0];
    const float* x2 = (const float*)d_in[1];
    const float* Wq = (const float*)d_in[2];
    const float* bq = (const float*)d_in[3];
    const float* Wk = (const float*)d_in[4];
    const float* bk = (const float*)d_in[5];
    const float* Wv = (const float*)d_in[6];
    const float* bv = (const float*)d_in[7];
    float* out = (float*)d_out;

    // Workspace: 3 x B*N*C bf16 (48 MB) + 3 x 32768 u32 swizzled weights
    __bf16* Qt = (__bf16*)d_ws;
    __bf16* Kt = Qt + (size_t)B_ * N_ * C_;
    __bf16* Vm = Kt + (size_t)B_ * N_ * C_;
    unsigned int* WswQ = (unsigned int*)(Vm + (size_t)B_ * N_ * C_);
    unsigned int* WswK = WswQ + 32768;
    unsigned int* WswV = WswK + 32768;

    wswz_kernel<<<256, 128, 0, stream>>>(Wq, WswQ);
    wswz_kernel<<<256, 128, 0, stream>>>(Wk, WswK);
    wswz_kernel<<<256, 128, 0, stream>>>(Wv, WswV);

    dim3 pg(N_ / 16, B_);
    proj_kernel<1><<<pg, 32, 0, stream>>>(x1, WswQ, bq, Qt);
    proj_kernel<1><<<pg, 32, 0, stream>>>(x2, WswK, bk, Kt);
    proj_kernel<0><<<pg, 32, 0, stream>>>(x2, WswV, bv, Vm);

    dim3 ag(N_ / 64, B_);
    attn_kernel<<<ag, 128, 0, stream>>>(Qt, Kt, Vm, out);
}
